// SingleRenderer_43069932044933
// MI455X (gfx1250) — matrix operand order repaired and checked by reading, not compiled, not run on hardware
//
#include <hip/hip_runtime.h>
#include <hip/hip_bf16.h>
#include <math.h>

typedef __attribute__((ext_vector_type(16))) _Float16 v16h;
typedef __attribute__((ext_vector_type(8)))  _Float16 v8h;
typedef __attribute__((ext_vector_type(8)))  float    v8f;

#define NRAYS          32768
#define NS_FINAL       128
#define RAYS_PER_BLOCK 32
#define THREADS        64

__device__ __forceinline__ float sigm(float x) { return 1.0f / (1.0f + __expf(-x)); }

// Wave-local LDS ordering: LDS ops from one wave execute in order (ISA 7.3),
// so cross-lane RAW within a wave only needs a compiler fence; s_wait_dscnt 0
// is belt-and-braces so the loads issue after stores have fully retired.
__device__ __forceinline__ void wave_lds_fence() {
    asm volatile("s_wait_dscnt 0" ::: "memory");
}

// Scalar SDF MLP: base(|p|-0.7) + 0.05*(relu(p@W1+b1)@W2 + b2)
__device__ __forceinline__ float sdf_eval(const float* sW1, const float* sB1,
                                          const float* sW2, float b2v,
                                          float px, float py, float pz) {
    float base = sqrtf(px * px + py * py + pz * pz) - 0.7f;
    float corr = b2v;
#pragma unroll 8
    for (int j = 0; j < 32; ++j) {
        float h = fmaf(px, sW1[j], fmaf(py, sW1[32 + j], fmaf(pz, sW1[64 + j], sB1[j])));
        corr = fmaf(fmaxf(h, 0.0f), sW2[j], corr);
    }
    return base + 0.05f * corr;
}

__global__ __launch_bounds__(THREADS) void neus_render_kernel(
    const float* __restrict__ rays_o, const float* __restrict__ rays_d,
    const float* __restrict__ w1, const float* __restrict__ b1,
    const float* __restrict__ w2, const float* __restrict__ b2,
    const float* __restrict__ rw1, const float* __restrict__ rb1,
    const float* __restrict__ rw2, const float* __restrict__ rb2,
    const float* __restrict__ s_ptr, float* __restrict__ out)
{
    __shared__ float sW1[96], sB1[32], sW2[32], sRW1[192], sRB1[32], sRW2[96];
    __shared__ float sB2, sRB2[3], sS;
    __shared__ float d_lds[RAYS_PER_BLOCK][NS_FINAL];
    __shared__ float sdf_lds[RAYS_PER_BLOCK][NS_FINAL];
    __shared__ float w_lds[RAYS_PER_BLOCK][NS_FINAL];
    __shared__ float ray_od[RAYS_PER_BLOCK][6];
    __shared__ __align__(16) _Float16 h_tile[2][16][32];

    const int tid = threadIdx.x;

    // ---- cooperative parameter load ----
    for (int i = tid; i < 96; i += THREADS)  sW1[i] = w1[i];
    for (int i = tid; i < 32; i += THREADS)  { sB1[i] = b1[i]; sW2[i] = w2[i]; sRB1[i] = rb1[i]; }
    for (int i = tid; i < 192; i += THREADS) sRW1[i] = rw1[i];
    for (int i = tid; i < 96; i += THREADS)  sRW2[i] = rw2[i];
    if (tid == 0) { sB2 = b2[0]; sS = s_ptr[0]; }
    if (tid < 3) sRB2[tid] = rb2[tid];
    __syncthreads();

    // ================= Phase A0: initial 64 coarse samples, 2 threads per ray =================
    {
        const int rloc = tid & 31;          // local ray
        const int half = tid >> 5;          // 0: even samples, 1: odd samples
        const int ray = blockIdx.x * RAYS_PER_BLOCK + rloc;
        float ox = rays_o[ray * 3 + 0], oy = rays_o[ray * 3 + 1], oz = rays_o[ray * 3 + 2];
        float dx = rays_d[ray * 3 + 0], dy = rays_d[ray * 3 + 1], dz = rays_d[ray * 3 + 2];
        float inv = 1.0f / sqrtf(dx * dx + dy * dy + dz * dz);
        dx *= inv; dy *= inv; dz *= inv;
        if (half == 0) {
            ray_od[rloc][0] = ox; ray_od[rloc][1] = oy; ray_od[rloc][2] = oz;
            ray_od[rloc][3] = dx; ray_od[rloc][4] = dy; ray_od[rloc][5] = dz;
        }
        const float mid = -(ox * dx + oy * dy + oz * dz);
        const float nearv = mid - 1.0f, farv = mid + 1.0f;
        for (int i = half; i < 64; i += 2) {
            float t = (float)i * (1.0f / 63.0f);
            float dv = nearv * (1.0f - t) + farv * t;
            d_lds[rloc][i] = dv;
            sdf_lds[rloc][i] = sdf_eval(sW1, sB1, sW2, sB2,
                                        ox + dv * dx, oy + dv * dy, oz + dv * dz);
        }
    }
    __syncthreads();

    // ================= Phase A1: sequential hierarchical upsampling (1 thread / ray) =========
    if (tid < RAYS_PER_BLOCK) {
        const float ox = ray_od[tid][0], oy = ray_od[tid][1], oz = ray_od[tid][2];
        const float dx = ray_od[tid][3], dy = ray_od[tid][4], dz = ray_od[tid][5];
        float* drow = d_lds[tid];
        float* srow = sdf_lds[tid];
        float* wrow = w_lds[tid];

        int n = 64;
        float carr[112];
        float fd[16], fs[16];
        for (int it = 0; it < 4; ++it) {
            const float si = 256.0f * (float)(1 << it);
            float T = 1.0f, pdfsum = 0.0f, raw_prev = 0.0f;
            for (int i = 0; i < n - 1; ++i) {
                float ps = srow[i], ns = srow[i + 1];
                float pzv = drow[i], nzv = drow[i + 1];
                float msdf = 0.5f * (ps + ns);
                float raw = (ns - ps) / (nzv - pzv + 1e-5f);
                float pd = (i == 0) ? 0.0f : raw_prev;
                float dvv = fminf(pd, raw);
                dvv = fminf(fmaxf(dvv, -10.0f), 0.0f);
                raw_prev = raw;
                float dist = nzv - pzv;
                float pe = msdf - dvv * dist * 0.5f;
                float ne = msdf + dvv * dist * 0.5f;
                float pc = sigm(pe * si), nc = sigm(ne * si);
                float alpha = (pc - nc + 1e-5f) / (pc + 1e-5f);
                float w = alpha * T;
                T *= (1.0f - alpha + 1e-10f);
                wrow[i] = w;
                pdfsum += w + 1e-5f;
            }
            const float invsum = 1.0f / pdfsum;
            carr[0] = 0.0f;
            float cum = 0.0f;
            for (int i = 0; i < n - 1; ++i) { cum += (wrow[i] + 1e-5f) * invsum; carr[i + 1] = cum; }
            for (int j = 0; j < 16; ++j) {
                float u = (float)j * (1.0f / 15.0f);
                int idx = 0;
                while (idx < n && carr[idx] <= u) ++idx;             // searchsorted 'right'
                int be = idx - 1; if (be < 0) be = 0; if (be > n - 1) be = n - 1;
                int ab = idx;     if (ab > n - 1) ab = n - 1;
                float c0 = carr[be], c1 = carr[ab];
                float den = c1 - c0; if (den < 1e-5f) den = 1.0f;
                float tt = (u - c0) / den;
                float b0 = drow[be], b1v = drow[ab];
                float dfine = b0 + tt * (b1v - b0);
                fd[j] = dfine;
                fs[j] = sdf_eval(sW1, sB1, sW2, sB2,
                                 ox + dfine * dx, oy + dfine * dy, oz + dfine * dz);
            }
            // stable backward merge of sorted (drow[0..n-1]) with sorted (fd[0..15])
            int i = n - 1, jj = 15, k = n + 15;
            while (jj >= 0) {
                if (i >= 0 && drow[i] > fd[jj]) { drow[k] = drow[i]; srow[k] = srow[i]; --i; }
                else                            { drow[k] = fd[jj];  srow[k] = fs[jj];  --jj; }
                --k;
            }
            n += 16;
        }
    }
    __syncthreads();

    // ================= Phase B: weights scan + WMMA radiance MLP =================
    const int wv = tid >> 5;           // wave id (0..1)
    const int l  = tid & 31;           // lane
    const int lm = l & 15;
    const bool hiHalf = (l >= 16);

    // per-ray transmittance scan (lanes 0-15 of each wave, one ray per lane)
    if (l < 16) {
        const int row = wv * 16 + l;
        const int ray = blockIdx.x * RAYS_PER_BLOCK + row;
        const float s = sS;
        const float* srow = sdf_lds[row];
        const float* drow = d_lds[row];
        float* wrow = w_lds[row];
        float T = 1.0f, acc = 0.0f, dsum = 0.0f;
        float cp = sigm(srow[0] * s);
        for (int i = 0; i < 127; ++i) {
            float cn = sigm(srow[i + 1] * s);
            float alpha = fmaxf((cp - cn) / (cp + 1e-10f), 0.0f);
            float w = alpha * T;
            T *= (1.0f - alpha + 1e-10f);
            wrow[i] = w;
            float dm = 0.5f * (drow[i] + drow[i + 1]);
            acc += w; dsum += w * dm;
            cp = cn;
        }
        out[ray * 5 + 3] = dsum / (acc + 1e-10f);
        out[ray * 5 + 4] = acc;
    }
    __syncthreads();

    // ---- B-matrix tiles (per-lane registers) for the radiance MLP ----
    // Layer1 B (32x16 f16, K rows 0..5 = rw1, K=6 = bias): lanes 0-15 hold K0..15 of column lm.
    v16h bw1lo = {}; v16h bw1hi = {}; v16h bw2 = {};
    if (!hiHalf) {
#pragma unroll
        for (int k = 0; k < 6; ++k) {
            bw1lo[k] = (_Float16)sRW1[k * 32 + lm];
            bw1hi[k] = (_Float16)sRW1[k * 32 + 16 + lm];
        }
        bw1lo[6] = (_Float16)sRB1[lm];
        bw1hi[6] = (_Float16)sRB1[16 + lm];
    }
    // Layer2 B (32x16, columns 0..2 = rw2): lanes 0-15 hold K0..7,16..23, lanes 16-31 K8..15,24..31.
    if (lm < 3) {
        const int kb = hiHalf ? 8 : 0;
#pragma unroll
        for (int e = 0; e < 16; ++e) bw2[e] = (_Float16)sRW2[(kb + e + (e >= 8 ? 8 : 0)) * 3 + lm];
    }

    for (int rr = 0; rr < 16; ++rr) {
        const int row = wv * 16 + rr;
        const int ray = blockIdx.x * RAYS_PER_BLOCK + row;
        const float ox = ray_od[row][0], oy = ray_od[row][1], oz = ray_od[row][2];
        const float ddx = ray_od[row][3], ddy = ray_od[row][4], ddz = ray_od[row][5];
        float rgb_part = 0.0f;

        for (int t = 0; t < 8; ++t) {
            // A tile: 16 mid-points x K(=7 used). Lanes 16-31 carry zero K-slices.
            v16h a = {};
            if (!hiHalf) {
                const int m = t * 16 + lm;
                if (m < 127) {
                    float dm = 0.5f * (d_lds[row][m] + d_lds[row][m + 1]);
                    a[0] = (_Float16)(ox + dm * ddx);
                    a[1] = (_Float16)(oy + dm * ddy);
                    a[2] = (_Float16)(oz + dm * ddz);
                    a[3] = (_Float16)ddx;
                    a[4] = (_Float16)ddy;
                    a[5] = (_Float16)ddz;
                    a[6] = (_Float16)1.0f;
                }
            }
            v8f cz = {};
            v8f c0 = __builtin_amdgcn_wmma_f32_16x16x32_f16(false, a, false, bw1lo, (short)0, cz, false, false);
            v8f c1 = __builtin_amdgcn_wmma_f32_16x16x32_f16(false, a, false, bw1hi, (short)0, cz, false, false);

            // relu + relayout D -> A via per-wave LDS staging tile (wave-local ordering only)
            _Float16* ht = &h_tile[wv][0][0];
            const int rowbase = hiHalf ? 8 : 0;
#pragma unroll
            for (int r = 0; r < 8; ++r) ht[(rowbase + r) * 32 + lm]      = (_Float16)fmaxf(c0[r], 0.0f);
#pragma unroll
            for (int r = 0; r < 8; ++r) ht[(rowbase + r) * 32 + 16 + lm] = (_Float16)fmaxf(c1[r], 0.0f);
            wave_lds_fence();

            v16h a2;
            const int kk = hiHalf ? 8 : 0;
            ((v8h*)&a2)[0] = *(const v8h*)&ht[lm * 32 + kk];
            ((v8h*)&a2)[1] = *(const v8h*)&ht[lm * 32 + 16 + kk];
            wave_lds_fence();   // keep next iteration's stores behind these loads

            v8f d2 = __builtin_amdgcn_wmma_f32_16x16x32_f16(false, a2, false, bw2, (short)0, cz, false, false);

            // lanes {0,1,2} hold rows 0..7, lanes {16,17,18} rows 8..15 of channel lm
            if (lm < 3) {
#pragma unroll
                for (int r = 0; r < 8; ++r) {
                    int m = t * 16 + r + (hiHalf ? 8 : 0);
                    if (m < 127) {
                        float rad = sigm(d2[r] + sRB2[lm]);
                        rgb_part += w_lds[row][m] * rad;
                    }
                }
            }
        }
        float other = __shfl_xor(rgb_part, 16, 32);
        rgb_part += other;
        if (!hiHalf && lm < 3) out[ray * 5 + lm] = rgb_part;
    }
}

extern "C" void kernel_launch(void* const* d_in, const int* in_sizes, int n_in,
                              void* d_out, int out_size, void* d_ws, size_t ws_size,
                              hipStream_t stream) {
    (void)in_sizes; (void)n_in; (void)out_size; (void)d_ws; (void)ws_size;
    const float* rays_o = (const float*)d_in[0];
    const float* rays_d = (const float*)d_in[1];
    const float* w1  = (const float*)d_in[2];
    const float* b1  = (const float*)d_in[3];
    const float* w2  = (const float*)d_in[4];
    const float* b2  = (const float*)d_in[5];
    const float* rw1 = (const float*)d_in[6];
    const float* rb1 = (const float*)d_in[7];
    const float* rw2 = (const float*)d_in[8];
    const float* rb2 = (const float*)d_in[9];
    const float* s   = (const float*)d_in[10];
    float* out = (float*)d_out;

    dim3 grid(NRAYS / RAYS_PER_BLOCK);   // 1024 blocks
    dim3 block(THREADS);                 // 64 threads = 2 wave32
    neus_render_kernel<<<grid, block, 0, stream>>>(rays_o, rays_d, w1, b1, w2, b2,
                                                   rw1, rb1, rw2, rb2, s, out);
}
